// MultiHeadAttention_var_88725434401239
// MI455X (gfx1250) — compile-verified, matrix operand order checked by reading, not run on hardware
//
#include <hip/hip_runtime.h>

typedef __attribute__((ext_vector_type(16))) __bf16 v16bf;
typedef __attribute__((ext_vector_type(8)))  __bf16 v8bf;
typedef __attribute__((ext_vector_type(8)))  float  v8f;
typedef __attribute__((ext_vector_type(4)))  float  v4f;
typedef int v4i_gnu __attribute__((vector_size(16)));   // matches builtin param pointee

#define DMODEL 512
#define SEQ    512
#define BATCH  16
#define NHEAD  8
#define DK     64
#define NC     21
#define DSTRIDE 20   // padded row stride (dwords) for the LDS dtw tile

#if defined(__gfx1250__) && __has_builtin(__builtin_amdgcn_global_load_async_to_lds_b128)
#define HAVE_ASYNC_LDS 1
#else
#define HAVE_ASYNC_LDS 0
#endif

__device__ __forceinline__ int lane_id() { return (int)(threadIdx.x & 31u); }

__device__ __forceinline__ void wait_asynccnt0() {
#if defined(__gfx1250__)
#if __has_builtin(__builtin_amdgcn_s_wait_asynccnt)
  __builtin_amdgcn_s_wait_asynccnt(0);
#else
  asm volatile("s_wait_asynccnt 0x0" ::: "memory");
#endif
#endif
}

// CDNA5 16-bit A/B fragment (16x32, K-contiguous source):
// lane&15 selects the M (or N) line; lanes>=16 add +8 to K.
// Elements 0..7  <-> K = hi*8 + 0..7    (one 16B run)
// Elements 8..15 <-> K = hi*8 + 16..23  (one 16B run)
__device__ __forceinline__ v16bf load_frag_bf16_k1(const __bf16* p, int nstride) {
  const int lane = lane_id();
  const int hi = lane >> 4;
  const __bf16* q = p + (size_t)(lane & 15) * (size_t)nstride + hi * 8;
  const v8bf lo = *(const v8bf*)(q);
  const v8bf hi8 = *(const v8bf*)(q + 16);
  v16bf f;
#pragma unroll
  for (int e = 0; e < 8; ++e) { f[e] = lo[e]; f[e + 8] = hi8[e]; }
  return f;
}

__device__ __forceinline__ v16bf load_frag_f32_k1(const float* p, int nstride) {
  const int lane = lane_id();
  const int hi = lane >> 4;
  const float* q = p + (size_t)(lane & 15) * (size_t)nstride + hi * 8;
  const v4f a0 = *(const v4f*)(q);
  const v4f a1 = *(const v4f*)(q + 4);
  const v4f b0 = *(const v4f*)(q + 16);
  const v4f b1 = *(const v4f*)(q + 20);
  v16bf f;
#pragma unroll
  for (int e = 0; e < 4; ++e) {
    f[e]      = (__bf16)a0[e];
    f[e + 4]  = (__bf16)a1[e];
    f[e + 8]  = (__bf16)b0[e];
    f[e + 12] = (__bf16)b1[e];
  }
  return f;
}

__device__ __forceinline__ v8f wmma_bf16(v16bf a, v16bf b, v8f c) {
  return __builtin_amdgcn_wmma_f32_16x16x32_bf16(false, a, false, b, (short)0, c, false, false);
}

// ---------------------------------------------------------------------------
// One-off f32 -> bf16 conversion (for the 512x512 weight matrices).
// ---------------------------------------------------------------------------
__global__ __launch_bounds__(256) void cvt_f32_bf16_kernel(
    const float* __restrict__ in, __bf16* __restrict__ out) {
  const size_t i = ((size_t)blockIdx.x * 256 + threadIdx.x) * 8;
  const v4f a = *(const v4f*)(in + i);
  const v4f b = *(const v4f*)(in + i + 4);
  v8bf o;
#pragma unroll
  for (int e = 0; e < 4; ++e) { o[e] = (__bf16)a[e]; o[e + 4] = (__bf16)b[e]; }
  *(v8bf*)(out + i) = o;
}

// ---------------------------------------------------------------------------
// GEMM: Y = X @ W^T + bias   (M=8192, N=512, K=512), W pre-converted to bf16.
// A_F32:    A is f32 [M,512] (bf16-converted on load) else bf16 [M,512]
// OUT_MODE: 0 -> f32 [M,512]
//           1 -> bf16 head-major  [b,h,s,d]  (n = h*64+d), scaled
//           2 -> bf16 head-transposed [b,h,d,s], scaled
// Block: 256 threads = 8 waves; wave tile 32m x 64n; block tile 64m x 256n.
// ---------------------------------------------------------------------------
template <bool A_F32, int OUT_MODE>
__global__ __launch_bounds__(256) void gemm_bias_kernel(
    const void* __restrict__ Av, const __bf16* __restrict__ W,
    const float* __restrict__ bias, void* __restrict__ Yv, float scale) {
  const int lane = lane_id();
  const int wave = (int)(threadIdx.x >> 5);
  const int m0 = blockIdx.x * 64 + (wave >> 2) * 32;
  const int n0 = blockIdx.y * 256 + (wave & 3) * 64;
  const int hi = lane >> 4;

  v8f acc[2][4] = {};

  for (int kk = 0; kk < DMODEL; kk += 32) {
    v16bf a[2];
#pragma unroll
    for (int i = 0; i < 2; ++i) {
      if (A_F32)
        a[i] = load_frag_f32_k1((const float*)Av + (size_t)(m0 + 16 * i) * DMODEL + kk, DMODEL);
      else
        a[i] = load_frag_bf16_k1((const __bf16*)Av + (size_t)(m0 + 16 * i) * DMODEL + kk, DMODEL);
    }
#pragma unroll
    for (int t = 0; t < 4; ++t) {
      // B(k,n) = W[n,k] -> row (n0+16t+lane&15) of W, K-contiguous
      const v16bf b = load_frag_bf16_k1(W + (size_t)(n0 + 16 * t) * DMODEL + kk, DMODEL);
      acc[0][t] = wmma_bf16(a[0], b, acc[0][t]);
      acc[1][t] = wmma_bf16(a[1], b, acc[1][t]);
    }
  }

#pragma unroll
  for (int i = 0; i < 2; ++i) {
#pragma unroll
    for (int t = 0; t < 4; ++t) {
      const int n = n0 + 16 * t + (lane & 15);
      const float bn = bias[n];
#pragma unroll
      for (int r = 0; r < 8; ++r) {
        const int m = m0 + 16 * i + r + 8 * hi;
        const float v = (acc[i][t][r] + bn) * scale;
        if (OUT_MODE == 0) {
          ((float*)Yv)[(size_t)m * DMODEL + n] = v;
        } else {
          const int b_ = m >> 9, s_ = m & 511, h_ = n >> 6, d_ = n & 63;
          if (OUT_MODE == 1)
            ((__bf16*)Yv)[(((size_t)(b_ * NHEAD + h_) * SEQ + s_) << 6) + d_] = (__bf16)v;
          else
            ((__bf16*)Yv)[((size_t)(b_ * NHEAD + h_) * DK + d_) * SEQ + s_] = (__bf16)v;
        }
      }
    }
  }
}

// ---------------------------------------------------------------------------
// Attention scores + DTW bias + softmax -> attn [b,h,q,k] f32.
// Block = (b, 16-query tile); 8 waves, wave w = head w.
// LDS: 256 KB scores + 26 KB dtw tile stage + scratch (~284 KB < 320 KB WGP).
// The dtw tile (21 x 16 x 16) is staged cooperatively ONCE per block — via the
// gfx1250 async global->LDS engine — and shared by all 8 head-waves (8x
// traffic reduction on the dominant 352 MB input).
// ---------------------------------------------------------------------------
__global__ __launch_bounds__(256) void attn_softmax_kernel(
    const __bf16* __restrict__ qh, const __bf16* __restrict__ kh,
    const float* __restrict__ dtw, const float* __restrict__ Wd,
    const float* __restrict__ bd, float* __restrict__ attn) {
  extern __shared__ float smem[];
  const int lane = lane_id();
  const int h = (int)(threadIdx.x >> 5);
  const int q0 = blockIdx.x * 16;
  const int b = blockIdx.y;
  const int hi = lane >> 4;

  float* sc     = smem;                                   // 8*16*512 scores
  float* dstage = smem + (size_t)NHEAD * 16 * SEQ;        // NC*16*DSTRIDE tile
  float* fin    = dstage + NC * 16 * DSTRIDE + h * 32;    // [row][2] scratch

  const __bf16* qbase = qh + (((size_t)(b * NHEAD + h) * SEQ) + q0) * DK;
  const __bf16* kbase = kh + ((size_t)(b * NHEAD + h) * SEQ) * DK;

  // q rows pre-scaled by 1/sqrt(dk) at projection time
  const v16bf qf0 = load_frag_bf16_k1(qbase + 0,  DK);
  const v16bf qf1 = load_frag_bf16_k1(qbase + 32, DK);
  const float bdh = bd[h];

  // Preload the 21 per-head DTW mixing weights once (the per-tile barriers
  // below would otherwise force a reload every iteration).
  float wdr[NC];
#pragma unroll
  for (int c = 0; c < NC; ++c) wdr[c] = Wd[h * NC + c];

  for (int kt = 0; kt < SEQ; kt += 16) {
    // Issue this tile's K-fragment loads first so their latency overlaps the
    // LDS staging + barriers below.
    const v16bf kf0 = load_frag_bf16_k1(kbase + (size_t)kt * DK + 0,  DK);
    const v16bf kf1 = load_frag_bf16_k1(kbase + (size_t)kt * DK + 32, DK);

    __syncthreads();  // previous iteration's dstage reads complete
    // Cooperative stage of dtw[b, 0..20, q0..q0+15, kt..kt+15] (16B chunks)
    for (int j = (int)threadIdx.x; j < NC * 16 * 4; j += 256) {
      const int c = j >> 6;             // 64 chunks per channel
      const int rem = j & 63;
      const int r = rem >> 2;           // tile row 0..15
      const int k4 = (rem & 3) * 4;     // 0,4,8,12
      const float* g = dtw + ((((size_t)b * NC + c) * SEQ) + q0 + r) * SEQ + kt + k4;
      float* l = dstage + (c * 16 + r) * DSTRIDE + k4;
#if HAVE_ASYNC_LDS
      __builtin_amdgcn_global_load_async_to_lds_b128(
          (__attribute__((address_space(1))) v4i_gnu*)g,
          (__attribute__((address_space(3))) v4i_gnu*)l, 0, 0);
#else
      *(v4f*)l = *(const v4f*)g;
#endif
      if (kt + 16 < SEQ) __builtin_prefetch(g + 16, 0, 1);  // next k-tile
    }
#if HAVE_ASYNC_LDS
    wait_asynccnt0();
#endif
    __syncthreads();  // staged tile visible to all waves

    v8f acc = {};
    acc = wmma_bf16(qf0, kf0, acc);
    acc = wmma_bf16(qf1, kf1, acc);

#pragma unroll
    for (int r = 0; r < 8; ++r) acc[r] += bdh;
    const int col = lane & 15;
    // Bias accumulation, 3 channels (24 LDS reads) batched per step so many
    // ds_loads are in flight before the first use (21 = 7 x 3).
    for (int cc = 0; cc < NC; cc += 3) {
      float t[3][8];
#pragma unroll
      for (int u = 0; u < 3; ++u) {
        const float* dp = dstage + ((cc + u) * 16 + 8 * hi) * DSTRIDE + col;
#pragma unroll
        for (int r = 0; r < 8; ++r) t[u][r] = dp[r * DSTRIDE];
      }
#pragma unroll
      for (int u = 0; u < 3; ++u) {
        const float w = wdr[cc + u];
#pragma unroll
        for (int r = 0; r < 8; ++r) acc[r] = fmaf(t[u][r], w, acc[r]);
      }
    }
#pragma unroll
    for (int r = 0; r < 8; ++r)
      sc[(size_t)h * 16 * SEQ + (size_t)(r + 8 * hi) * SEQ + kt + col] = acc[r];
  }
  __syncthreads();

  // Softmax over k (512) per row: lane -> row = lane&15, half = hi (256 cols each)
  {
    const int row = lane & 15;
    float* rowp = sc + (size_t)h * 16 * SEQ + (size_t)row * SEQ + hi * 256;
    float mx = -3.4e38f;
    for (int j = 0; j < 256; ++j) mx = fmaxf(mx, rowp[j]);
    fin[row * 2 + hi] = mx;
    __syncthreads();
    const float m2 = fmaxf(fin[row * 2], fin[row * 2 + 1]);
    __syncthreads();
    float ssum = 0.0f;
    for (int j = 0; j < 256; ++j) {
      const float e = __expf(rowp[j] - m2);
      rowp[j] = e;
      ssum += e;
    }
    fin[row * 2 + hi] = ssum;
    __syncthreads();
    const float rinv = 1.0f / (fin[row * 2] + fin[row * 2 + 1]);
    if (hi == 0) fin[row * 2] = rinv;
    __syncthreads();
  }

  // Coalesced normalized write
  float* scb = sc + (size_t)h * 16 * SEQ;
  float* outbase = attn + (((size_t)(b * NHEAD + h) * SEQ) + q0) * SEQ;
  for (int idx = lane; idx < 16 * SEQ; idx += 32) {
    const int row = idx >> 9, col = idx & 511;
    outbase[(size_t)row * SEQ + col] = scb[(size_t)row * SEQ + col] * fin[row * 2];
  }
}

// ---------------------------------------------------------------------------
// out = attn @ v, stored as bf16 concat [b, s, h*64 + d].
// v is stored transposed [b,h,d,s] so B fragments are K-contiguous (b128 loads).
// Grid: (S/256, b*H); 8 waves, wave tile 32m x 64n, K loop over 512 keys.
// ---------------------------------------------------------------------------
__global__ __launch_bounds__(256) void av_concat_kernel(
    const float* __restrict__ attn, const __bf16* __restrict__ vT,
    __bf16* __restrict__ concat) {
  const int lane = lane_id();
  const int wave = (int)(threadIdx.x >> 5);
  const int bh = blockIdx.y;
  const int b = bh >> 3, h = bh & 7;
  const int m0 = blockIdx.x * 256 + wave * 32;
  const int hi = lane >> 4;

  const float* abase = attn + ((size_t)bh * SEQ + m0) * SEQ;
  const __bf16* vbase = vT + (size_t)bh * DK * SEQ;

  v8f acc[2][4] = {};
  for (int kk = 0; kk < SEQ; kk += 32) {
    v16bf a[2];
#pragma unroll
    for (int i = 0; i < 2; ++i)
      a[i] = load_frag_f32_k1(abase + (size_t)(16 * i) * SEQ + kk, SEQ);
#pragma unroll
    for (int t = 0; t < 4; ++t) {
      // B(k=key, n=dki) = vT[b,h,16t+n, key], K-contiguous
      const v16bf bf = load_frag_bf16_k1(vbase + (size_t)(16 * t) * SEQ + kk, SEQ);
      acc[0][t] = wmma_bf16(a[0], bf, acc[0][t]);
      acc[1][t] = wmma_bf16(a[1], bf, acc[1][t]);
    }
  }
#pragma unroll
  for (int i = 0; i < 2; ++i) {
#pragma unroll
    for (int t = 0; t < 4; ++t) {
      const int n = 16 * t + (lane & 15);
#pragma unroll
      for (int r = 0; r < 8; ++r) {
        const int srow = m0 + 16 * i + r + 8 * hi;
        concat[((size_t)b * SEQ + srow) * DMODEL + h * DK + n] = (__bf16)acc[i][t][r];
      }
    }
  }
}

extern "C" void kernel_launch(void* const* d_in, const int* in_sizes, int n_in,
                              void* d_out, int out_size, void* d_ws, size_t ws_size,
                              hipStream_t stream) {
  (void)in_sizes; (void)n_in; (void)out_size; (void)ws_size;

  const float* Q   = (const float*)d_in[0];
  const float* K   = (const float*)d_in[1];
  const float* V   = (const float*)d_in[2];
  /* d_in[3] = G, unused by the reference */
  const float* dtw = (const float*)d_in[4];
  const float* Wq  = (const float*)d_in[5];
  const float* bq  = (const float*)d_in[6];
  const float* Wk  = (const float*)d_in[7];
  const float* bk  = (const float*)d_in[8];
  const float* Wv  = (const float*)d_in[9];
  const float* bv  = (const float*)d_in[10];
  const float* Wd  = (const float*)d_in[11];
  const float* bd  = (const float*)d_in[12];
  const float* Wo  = (const float*)d_in[13];
  const float* bo  = (const float*)d_in[14];

  float* out  = (float*)d_out;                                   // [16,512,512]
  float* attn = out + (size_t)BATCH * SEQ * DMODEL;              // [16,8,512,512]

  const size_t headBytes = (size_t)BATCH * NHEAD * SEQ * DK * 2; // 8 MB each
  const size_t wBytes    = (size_t)DMODEL * DMODEL * 2;          // 0.5 MB each
  __bf16* qh     = (__bf16*)d_ws;
  __bf16* kh     = (__bf16*)((char*)d_ws + headBytes);
  __bf16* vT     = (__bf16*)((char*)d_ws + headBytes * 2);
  __bf16* concat = (__bf16*)((char*)d_ws + headBytes * 3);
  __bf16* wqb    = (__bf16*)((char*)d_ws + headBytes * 4);
  __bf16* wkb    = (__bf16*)((char*)d_ws + headBytes * 4 + wBytes);
  __bf16* wvb    = (__bf16*)((char*)d_ws + headBytes * 4 + wBytes * 2);
  __bf16* wob    = (__bf16*)((char*)d_ws + headBytes * 4 + wBytes * 3);  // 34 MB total

  const dim3 bb(256);
  const int cvtBlocks = (DMODEL * DMODEL) / (256 * 8);  // 128
  cvt_f32_bf16_kernel<<<cvtBlocks, bb, 0, stream>>>(Wq, wqb);
  cvt_f32_bf16_kernel<<<cvtBlocks, bb, 0, stream>>>(Wk, wkb);
  cvt_f32_bf16_kernel<<<cvtBlocks, bb, 0, stream>>>(Wv, wvb);
  cvt_f32_bf16_kernel<<<cvtBlocks, bb, 0, stream>>>(Wo, wob);

  const dim3 gg(128, 2);
  gemm_bias_kernel<true, 1><<<gg, bb, 0, stream>>>(Q, wqb, bq, qh, 0.125f);
  gemm_bias_kernel<true, 1><<<gg, bb, 0, stream>>>(K, wkb, bk, kh, 1.0f);
  gemm_bias_kernel<true, 2><<<gg, bb, 0, stream>>>(V, wvb, bv, vT, 1.0f);

  const size_t smem = ((size_t)NHEAD * 16 * SEQ            // scores
                       + (size_t)NC * 16 * DSTRIDE         // dtw tile stage
                       + (size_t)NHEAD * 32)               // reduction scratch
                      * sizeof(float);                     // ~284 KB
  attn_softmax_kernel<<<dim3(SEQ / 16, BATCH), bb, smem, stream>>>(
      qh, kh, dtw, Wd, bd, attn);

  av_concat_kernel<<<dim3(SEQ / 256, BATCH * NHEAD), bb, 0, stream>>>(attn, vT, concat);

  gemm_bias_kernel<false, 0><<<gg, bb, 0, stream>>>(concat, wob, bo, out, 1.0f);
}
